// MergedFFNExpert_61933428416507
// MI455X (gfx1250) — matrix operand order repaired and checked by reading, not compiled, not run on hardware
//
#include <hip/hip_runtime.h>

// Fused per-expert FFN: out = relu(X * W1) * W2, bf16 WMMA with f32 accumulate.
// X is [E][T=65536][64] contiguous, W1 [E][64][256], W2 [E][256][64], all fp32.
//
// GEMM1 is computed transposed (h^T = W1^T * x^T) so its C-fragment layout is
// bit-identical to the A-fragment layout GEMM2 needs: the intermediate h never
// leaves VGPRs. An opaque (asm-laundered) zero offset, regenerated inside the
// persistent tile loop, keeps the LDS weight-fragment loads inside the loop
// (prevents LICM hoist -> scratch spills) while still lowering to ds_load_b128.

typedef __attribute__((ext_vector_type(16))) __bf16       v16bf;
typedef __attribute__((ext_vector_type(8)))  float        v8f;
typedef __attribute__((ext_vector_type(4)))  float        v4f;
typedef __attribute__((ext_vector_type(4)))  unsigned int v4u;

#define DD   64      // d_model
#define FF   256     // dim_feedforward
#define TILE 128     // tokens per workgroup iteration (8 waves x 16 tokens)
#define WGS_PER_EXPERT 64

__device__ __forceinline__ __bf16 bf(float f) { return (__bf16)f; }

// A-fragment: two 16B runs at p and p+16 elements (K and K+16 sub-runs).
__device__ __forceinline__ v16bf lds_frag_a(const __bf16* p) {
    union { v4u u[2]; v16bf v; } t;
    t.u[0] = *(const v4u*)(p);
    t.u[1] = *(const v4u*)(p + 16);
    return t.v;
}
// B-fragment: 16 contiguous bf16 (32B) at p.
__device__ __forceinline__ v16bf lds_frag_b(const __bf16* p) {
    union { v4u u[2]; v16bf v; } t;
    t.u[0] = *(const v4u*)(p);
    t.u[1] = *(const v4u*)(p + 8);
    return t.v;
}

__global__ __launch_bounds__(256)
void moe_ffn_kernel(const float* __restrict__ x,
                    const float* __restrict__ w1,
                    const float* __restrict__ w2,
                    float* __restrict__ out,
                    int tilesPerExpert, int wgsPerExpert)
{
    // LDS: only the (reused) expert weights, bf16, transposed to [n][k].
    __shared__ __bf16 sW1[FF * DD];   // W1^T [f][d] : GEMM1' A operand, 32 KB
    __shared__ __bf16 sW2[DD * FF];   // W2^T [n][k] : GEMM2  B operand, 32 KB

    const int tid  = threadIdx.x;
    const int lane = tid & 31;
    const int wave = tid >> 5;
    const int ln   = lane & 15;   // tile row/col index
    const int hv   = lane >> 4;   // half-wave (selects K sub-ranges)

    const int e  = blockIdx.x / wgsPerExpert;
    const int wg = blockIdx.x % wgsPerExpert;

    // ---- stage expert weights into LDS as bf16 (amortized over all tiles) ----
    {
        const float* w1e = w1 + (long)e * DD * FF;  // [d<64][f<256]
        const float* w2e = w2 + (long)e * FF * DD;  // [k<256][n<64]
        #pragma unroll 4
        for (int i = 0; i < (DD * FF) / 256; ++i) {
            int idx = i * 256 + tid;
            sW1[(idx & (FF - 1)) * DD + (idx >> 8)] = bf(w1e[idx]);
            sW2[(idx & (DD - 1)) * FF + (idx >> 6)] = bf(w2e[idx]);
        }
    }
    __syncthreads();

    const long expBase = (long)e * tilesPerExpert * TILE;

    for (int tile = wg; tile < tilesPerExpert; tile += wgsPerExpert) {
        const long tok0 = expBase + (long)tile * TILE + wave * 16;

        // Opaque zero offset: regenerated every iteration, invisible to LICM,
        // so the LDS fragment loads below cannot be hoisted out of this loop.
        int zoff = 0;
        asm volatile("" : "+v"(zoff));
        const __bf16* w1z = sW1 + zoff;   // still addrspace(3) -> ds_load
        const __bf16* w2z = sW2 + zoff;

        // ---- x^T as GEMM1' B operand: lane = token ln, K runs {8hv+0..7},{16+8hv+0..7} ----
        // B layout (K=32 step): elements 0-7 hold K=8hv..8hv+7, 8-15 hold K=16+8hv..16+8hv+7.
        const float* xrow = x + (tok0 + ln) * DD + hv * 8;
        __builtin_prefetch(xrow + (long)wgsPerExpert * TILE * DD, 0, 1);

        v16bf bx[2];
        #pragma unroll
        for (int s = 0; s < 2; ++s) {
            float fv[16];
            *(v4f*)(fv + 0)  = *(const v4f*)(xrow + s * 32 + 0);
            *(v4f*)(fv + 4)  = *(const v4f*)(xrow + s * 32 + 4);
            *(v4f*)(fv + 8)  = *(const v4f*)(xrow + s * 32 + 16);
            *(v4f*)(fv + 12) = *(const v4f*)(xrow + s * 32 + 20);
            v16bf t;
            #pragma unroll
            for (int i = 0; i < 16; ++i) t[i] = bf(fv[i]);
            bx[s] = t;
        }

        // ---- GEMM1': for F-tile pair (2j, 2j+1): C' fragments ARE the GEMM2
        //      A fragments for k-step j (identity layout mapping). ----
        v16bf ah[8];
        #pragma unroll
        for (int j = 0; j < 8; ++j) {
            v8f acc0 = {}, acc1 = {};
            #pragma unroll
            for (int s = 0; s < 2; ++s) {
                // A frag of W1^T (16-bit A layout): lane row f = 16*mt + ln,
                // K runs {32s+8hv+0..7} and {32s+16+8hv+0..7}, contiguous in sW1.
                v16bf A0 = lds_frag_a(w1z + (32 * j + ln) * DD + s * 32 + hv * 8);
                v16bf A1 = lds_frag_a(w1z + (32 * j + 16 + ln) * DD + s * 32 + hv * 8);
                acc0 = __builtin_amdgcn_wmma_f32_16x16x32_bf16(
                           false, A0, false, bx[s], (short)0, acc0, false, false);
                acc1 = __builtin_amdgcn_wmma_f32_16x16x32_bf16(
                           false, A1, false, bx[s], (short)0, acc1, false, false);
            }
            // relu + cvt in registers: no LDS round trip for h.
            v16bf t;
            #pragma unroll
            for (int i = 0; i < 8; ++i) {
                float v0 = acc0[i], v1 = acc1[i];
                t[i]     = bf(v0 > 0.f ? v0 : 0.f);
                t[i + 8] = bf(v1 > 0.f ? v1 : 0.f);
            }
            ah[j] = t;
        }

        // ---- GEMM2: out[16x64] = h[16x256] * W2[256x64] ----
        float* orow = out + tok0 * DD;
        #pragma unroll
        for (int nt = 0; nt < DD / 16; ++nt) {
            v8f acc = {};
            #pragma unroll
            for (int s = 0; s < FF / 32; ++s) {
                v16bf b = lds_frag_a(w2z + (nt * 16 + ln) * FF + s * 32 + hv * 8);
                acc = __builtin_amdgcn_wmma_f32_16x16x32_bf16(
                          false, ah[s], false, b, (short)0, acc, false, false);
            }
            #pragma unroll
            for (int r = 0; r < 8; ++r) {
                orow[(long)(r + hv * 8) * DD + nt * 16 + ln] = acc[r];
            }
        }
    }
}

extern "C" void kernel_launch(void* const* d_in, const int* in_sizes, int n_in,
                              void* d_out, int out_size, void* d_ws, size_t ws_size,
                              hipStream_t stream) {
    const float* x  = (const float*)d_in[0];
    const float* w1 = (const float*)d_in[1];
    const float* w2 = (const float*)d_in[2];
    float* out      = (float*)d_out;

    const int numExperts     = in_sizes[1] / (DD * FF);      // 8
    const int tokensTotal    = in_sizes[0] / DD;             // 524288
    const int tokensPerExp   = tokensTotal / numExperts;     // 65536
    const int tilesPerExpert = tokensPerExp / TILE;          // 512

    dim3 grid(numExperts * WGS_PER_EXPERT);                  // 512 persistent WGs
    dim3 block(256);
    moe_ffn_kernel<<<grid, block, 0, stream>>>(x, w1, w2, out,
                                               tilesPerExpert, WGS_PER_EXPERT);
}
